// MultiInferCNNModel_55825984913741
// MI455X (gfx1250) — matrix operand order repaired and checked by reading, not compiled, not run on hardware
//
#include <hip/hip_runtime.h>
#include <hip/hip_bf16.h>

#define BATCH 8
#define LSEQ  128
#define CNN   256
#define CLS   6
#define ATT   50
#define INCH  400
#define MROWS (BATCH*LSEQ*LSEQ)   /* 131072 */
#define KTILES 17                 /* ceil(530/32) -> K padded to 544 */

typedef _Float16 h16;
typedef _Float16 v16h __attribute__((ext_vector_type(16)));
typedef float    v8f  __attribute__((ext_vector_type(8)));

// ---------------------------------------------------------------- embeddings
__global__ void embed_k(const int* __restrict__ x, const float* __restrict__ gen,
                        const float* __restrict__ dom, float* __restrict__ xe) {
    int idx = blockIdx.x * blockDim.x + threadIdx.x;
    if (idx >= BATCH * INCH * LSEQ) return;
    int l = idx & (LSEQ - 1);
    int c = (idx >> 7) % INCH;
    int b = idx / (INCH * LSEQ);
    int tok = x[b * LSEQ + l];
    float v = (c < 300) ? gen[(size_t)tok * 300 + c]
                        : dom[(size_t)tok * 100 + (c - 300)];
    xe[idx] = v;   // layout (B, 400, L)
}

// ------------------------------------------------- conv1 (k5) || conv2 (k3)
__global__ void conv12_k(const float* __restrict__ xe,
                         const float* __restrict__ w1, const float* __restrict__ b1,
                         const float* __restrict__ w2, const float* __restrict__ b2,
                         float* __restrict__ out) {
    int idx = blockIdx.x * blockDim.x + threadIdx.x;
    if (idx >= BATCH * CNN * LSEQ) return;
    int l  = idx & (LSEQ - 1);
    int co = (idx >> 7) & (CNN - 1);
    int b  = idx >> 15;
    const float* xb = xe + (size_t)b * INCH * LSEQ;
    float acc;
    if (co < 128) {
        acc = b1[co];
        const float* wrow = w1 + (size_t)co * INCH * 5;
        for (int ci = 0; ci < INCH; ++ci) {
            const float* xc = xb + ci * LSEQ;
            const float* wk = wrow + ci * 5;
#pragma unroll
            for (int k = 0; k < 5; ++k) {
                int p = l + k - 2;
                if ((unsigned)p < (unsigned)LSEQ) acc += xc[p] * wk[k];
            }
        }
    } else {
        int c2 = co - 128;
        acc = b2[c2];
        const float* wrow = w2 + (size_t)c2 * INCH * 3;
        for (int ci = 0; ci < INCH; ++ci) {
            const float* xc = xb + ci * LSEQ;
            const float* wk = wrow + ci * 3;
#pragma unroll
            for (int k = 0; k < 3; ++k) {
                int p = l + k - 1;
                if ((unsigned)p < (unsigned)LSEQ) acc += xc[p] * wk[k];
            }
        }
    }
    out[idx] = fmaxf(acc, 0.f);
}

// ------------------------------------------------------ conv 256->256 k5 p2
__global__ void conv256_k(const float* __restrict__ xin, const float* __restrict__ w,
                          const float* __restrict__ bias, float* __restrict__ out) {
    int idx = blockIdx.x * blockDim.x + threadIdx.x;
    if (idx >= BATCH * CNN * LSEQ) return;
    int l  = idx & (LSEQ - 1);
    int co = (idx >> 7) & (CNN - 1);
    int b  = idx >> 15;
    const float* xb = xin + (size_t)b * CNN * LSEQ;
    const float* wrow = w + (size_t)co * CNN * 5;
    float acc = bias[co];
    for (int ci = 0; ci < CNN; ++ci) {
        const float* xc = xb + ci * LSEQ;
        const float* wk = wrow + ci * 5;
#pragma unroll
        for (int k = 0; k < 5; ++k) {
            int p = l + k - 2;
            if ((unsigned)p < (unsigned)LSEQ) acc += xc[p] * wk[k];
        }
    }
    out[idx] = fmaxf(acc, 0.f);
}

// ---------------------------------------------------- (B,C,L) -> (B,L,C)
__global__ void transpose_k(const float* __restrict__ xin, float* __restrict__ xout) {
    int idx = blockIdx.x * blockDim.x + threadIdx.x;
    if (idx >= BATCH * LSEQ * CNN) return;
    int c = idx & (CNN - 1);
    int l = (idx >> 8) & (LSEQ - 1);
    int b = idx >> 15;
    xout[idx] = xin[((size_t)b * CNN + c) * LSEQ + l];
}

// ------------------------------------------------------------- q/v proj
__global__ void qv_k(const float* __restrict__ xc,
                     const float* __restrict__ wq, const float* __restrict__ bq,
                     const float* __restrict__ wv, const float* __restrict__ bv,
                     float* __restrict__ q, float* __restrict__ v) {
    int idx = blockIdx.x * blockDim.x + threadIdx.x;
    if (idx >= BATCH * LSEQ * 64) return;
    int a  = idx & 63;
    int bl = idx >> 6;
    if (a >= ATT) return;
    const float* row = xc + (size_t)bl * CNN;
    float aq = bq[a], av = bv[a];
    for (int k = 0; k < CNN; ++k) {
        float xv = row[k];
        aq += xv * wq[k * ATT + a];
        av += xv * wv[k * ATT + a];
    }
    q[bl * ATT + a] = aq;
    v[bl * ATT + a] = av;
}

// ------------------------------------------------------------- attn scores
__global__ void scores_k(const float* __restrict__ q, const float* __restrict__ v,
                         const float* __restrict__ vv, const float* __restrict__ mask,
                         float* __restrict__ sc) {
    int idx = blockIdx.x * blockDim.x + threadIdx.x;
    if (idx >= MROWS) return;
    int j = idx & 127;
    int i = (idx >> 7) & 127;
    int b = idx >> 14;
    const float* qi = q + (size_t)(b * LSEQ + i) * ATT;
    const float* vj = v + (size_t)(b * LSEQ + j) * ATT;
    float s = 0.f;
    for (int a = 0; a < ATT; ++a) s += tanhf(qi[a] + vj[a]) * vv[a];
    if (mask[b * LSEQ + j] == 0.f) s = -1e9f;
    sc[idx] = s;
}

// -------------------------------------- softmax + merged + residual + f16
__global__ __launch_bounds__(128) void attn_merge_k(const float* __restrict__ sc,
                                                    const float* __restrict__ xcT,
                                                    const float* __restrict__ mask,
                                                    float* __restrict__ xcF,
                                                    h16* __restrict__ xcH) {
    __shared__ float attn[LSEQ];
    __shared__ float red[LSEQ];
    int bi = blockIdx.x;                 // (b,i)
    int b = bi >> 7, i = bi & 127;
    int tid = threadIdx.x;
    float s = sc[(size_t)bi * LSEQ + tid];
    red[tid] = s;
    __syncthreads();
    for (int st = 64; st > 0; st >>= 1) {
        if (tid < st) red[tid] = fmaxf(red[tid], red[tid + st]);
        __syncthreads();
    }
    float mx = red[0];
    __syncthreads();
    float e = __expf(s - mx);
    attn[tid] = e;
    red[tid]  = e;
    __syncthreads();
    for (int st = 64; st > 0; st >>= 1) {
        if (tid < st) red[tid] += red[tid + st];
        __syncthreads();
    }
    float inv = 1.f / red[0];
    float mk  = mask[b * LSEQ + i];
    for (int d = tid; d < CNN; d += 128) {
        float acc = 0.f;
        for (int j = 0; j < LSEQ; ++j)
            acc += attn[j] * xcT[((size_t)(b * LSEQ + j)) * CNN + d];
        float val = xcT[((size_t)(b * LSEQ + i)) * CNN + d] + mk * acc * inv;
        xcF[((size_t)(b * LSEQ + i)) * CNN + d] = val;
        xcH[((size_t)(b * LSEQ + i)) * CNN + d] = (h16)val;
    }
}

// --------------------------- ci = xc @ wc[:256], cj = xc @ wc[256:]  (B,L,6)
__global__ void cicj_k(const float* __restrict__ xcF, const float* __restrict__ wc,
                       float* __restrict__ ci, float* __restrict__ cj) {
    int idx = blockIdx.x * blockDim.x + threadIdx.x;
    if (idx >= BATCH * LSEQ * 8) return;
    int c  = idx & 7;
    int bl = idx >> 3;
    if (c >= CLS) return;
    const float* row = xcF + (size_t)bl * CNN;
    float a0 = 0.f, a1 = 0.f;
    for (int k = 0; k < CNN; ++k) {
        a0 += row[k] * wc[k * CLS + c];
        a1 += row[k] * wc[(256 + k) * CLS + c];
    }
    ci[bl * CLS + c] = a0;
    cj[bl * CLS + c] = a1;
}

// ------------------------------------------ logits0[b,i,j] = ci_i + cj_j + bc
__global__ void logits0_k(const float* __restrict__ ci, const float* __restrict__ cj,
                          const float* __restrict__ bc, float* __restrict__ out) {
    int idx = blockIdx.x * blockDim.x + threadIdx.x;
    if (idx >= MROWS * 8) return;
    int c = idx & 7;
    if (c >= CLS) return;
    int gm = idx >> 3;
    int j = gm & 127, i = (gm >> 7) & 127, b = gm >> 14;
    out[(size_t)gm * CLS + c] =
        ci[(b * LSEQ + i) * CLS + c] + cj[(b * LSEQ + j) * CLS + c] + bc[c];
}

// -------------------------------------------- feat0 = [xc_i | xc_j] as f16
__global__ void feat0_k(const h16* __restrict__ xcH, h16* __restrict__ feat) {
    int idx = blockIdx.x * blockDim.x + threadIdx.x;     // float4 units
    if (idx >= MROWS * 64) return;
    int n4 = idx & 63;
    int gm = idx >> 6;
    int j = gm & 127, i = (gm >> 7) & 127, b = gm >> 14;
    int n = n4 << 3;
    int srcRow = (n < 256) ? (b * LSEQ + i) : (b * LSEQ + j);
    const float4* src = (const float4*)(xcH + (size_t)srcRow * CNN + (n & 255));
    ((float4*)feat)[idx] = *src;
}

// ---------------- pre-pack wf (530x512 f32) into WMMA B-fragment layout f16
// bpack[kt][nt][lane][e]: lane -> N = nt*16 + lane%16 ; K = kt*32 + (lane/16)*16 + e
__global__ void packwf_k(const float* __restrict__ wf, h16* __restrict__ bpack) {
    int idx = blockIdx.x * blockDim.x + threadIdx.x;
    if (idx >= KTILES * 32 * 32) return;
    int lane = idx & 31;
    int nt   = (idx >> 5) & 31;
    int kt   = idx >> 10;
    int n     = nt * 16 + (lane & 15);
    int kbase = kt * 32 + (lane >> 4) * 16;
    __align__(16) h16 tmp[16];
#pragma unroll
    for (int e = 0; e < 16; ++e) {
        int K = kbase + e;
        tmp[e] = (K < 530) ? (h16)wf[(size_t)K * 512 + n] : (h16)0.f;
    }
    float4* dst = (float4*)(bpack + (size_t)idx * 16);
    dst[0] = ((float4*)tmp)[0];
    dst[1] = ((float4*)tmp)[1];
}

// --------------------------------------- lm = max(triu row/col maxes) (B,L,6)
__global__ void lm_k(const float* __restrict__ probs, float* __restrict__ lm) {
    int idx = blockIdx.x * blockDim.x + threadIdx.x;
    if (idx >= BATCH * LSEQ * 8) return;
    int c = idx & 7;
    if (c >= CLS) return;
    int t = (idx >> 3) & 127;
    int b = idx >> 10;
    float m1 = -INFINITY, m2 = -INFINITY;
    for (int i = 0; i < LSEQ; ++i) {
        float vA = (i <= t) ? probs[((size_t)(b * LSEQ + i) * LSEQ + t) * CLS + c] : 0.f;
        m1 = fmaxf(m1, vA);
        float vB = (i >= t) ? probs[((size_t)(b * LSEQ + t) * LSEQ + i) * CLS + c] : 0.f;
        m2 = fmaxf(m2, vB);
    }
    lm[(b * LSEQ + t) * CLS + c] = fmaxf(m1, m2);
}

// -------- tail tile (K = 512..543) per row: [lm_i(6) | lm_j(6) | probs(6) | 0]
__global__ void tail_k(const float* __restrict__ lm, const float* __restrict__ probs,
                       h16* __restrict__ tail) {
    int idx = blockIdx.x * blockDim.x + threadIdx.x;
    if (idx >= MROWS * 32) return;
    int kloc = idx & 31;
    int gm = idx >> 5;
    int j = gm & 127, i = (gm >> 7) & 127, b = gm >> 14;
    float val = 0.f;
    if (kloc < 6)       val = lm[(b * LSEQ + i) * CLS + kloc];
    else if (kloc < 12) val = lm[(b * LSEQ + j) * CLS + (kloc - 6)];
    else if (kloc < 18) val = probs[(size_t)gm * CLS + (kloc - 12)];
    tail[idx] = (h16)val;
}

// =============================== THE HOP GEMM (WMMA f16 -> f32) =============
// featOut[M,512] = A[M,544] @ wf_pack + bf.
// A tiles stream into double-buffered LDS via per-lane-scatter async DMA
// (GLOBAL_LOAD_ASYNC_TO_LDS_B128, ASYNCcnt), one tile kept in flight behind
// the WMMAs. Block: 256 thr = 8 waves (2M x 4N), tile 64x128, wave tile 32x32.
__global__ __launch_bounds__(256) void hop_gemm_k(
        const h16* __restrict__ featIn, h16* __restrict__ featOut,
        const h16* __restrict__ bpack,  const h16* __restrict__ tail,
        const float* __restrict__ bf) {
    __shared__ __align__(32) h16 ldsA[2][4 * 32 * 16];   // 2 x 4KB, fragment order
    const int tid  = threadIdx.x;
    const int lane = tid & 31;
    const int wave = tid >> 5;
    const int wm = wave >> 2;          // 0..1  (M group)
    const int wn = wave & 3;           // 0..3  (N group)
    const int M0 = blockIdx.y * 64;
    const int N0 = blockIdx.x * 128;

    // per-thread copy job: (row bm, K-octet q) -> one 16B fragment slot
    // fragment layout: half=(q>>1), e0=8*(q&1), lane=(bm&15)+16*half
    const int bm = tid >> 2;
    const int q  = tid & 3;
    const int gmB = M0 + bm;
    const int rgB   = bm >> 4;
    const int laneB = (bm & 15) + 16 * (q >> 1);
    const int e0    = (q & 1) * 8;
    const unsigned fragOff  = (unsigned)(((rgB * 32 + laneB) * 16 + e0) * sizeof(h16));
    const unsigned ldsBase0 = (unsigned)(uintptr_t)(void*)&ldsA[0][0] + fragOff;
    const unsigned ldsBase1 = (unsigned)(uintptr_t)(void*)&ldsA[1][0] + fragOff;
    const unsigned long long gFeat =
        (unsigned long long)(const void*)(featIn + (size_t)gmB * 512 + q * 8);
    const unsigned long long gTail =
        (unsigned long long)(const void*)(tail + (size_t)gmB * 32 + q * 8);

    const int nt0 = (N0 >> 4) + wn * 2;          // global N-tile index

    v8f acc00 = {}, acc01 = {}, acc10 = {}, acc11 = {};

#define ISSUE(kt) do {                                                        \
        unsigned long long ga_ = ((kt) < 16)                                  \
            ? (gFeat + (unsigned long long)(kt) * 64ull) : gTail;             \
        unsigned dst_ = (((kt) & 1) ? ldsBase1 : ldsBase0);                   \
        asm volatile("global_load_async_to_lds_b128 %0, %1, off"              \
                     :: "v"(dst_), "v"(ga_) : "memory");                      \
    } while (0)

    ISSUE(0);
#pragma unroll
    for (int kt = 0; kt < KTILES; ++kt) {
        if (kt + 1 < KTILES) {
            ISSUE(kt + 1);                                 // keep 1 tile in flight
            asm volatile("s_wait_asynccnt 0x1" ::: "memory");
        } else {
            asm volatile("s_wait_asynccnt 0x0" ::: "memory");
        }
        __syncthreads();                                   // tile kt visible to all

        if (kt + 2 < KTILES)                               // WGP-scope prefetch of B
            __builtin_prefetch(bpack + ((size_t)((kt + 2) * 32 + nt0) * 32 + lane) * 16, 0, 3);

        const h16* buf = ldsA[kt & 1];
        v16h a0 = *(const v16h*)&buf[((wm * 2 + 0) * 32 + lane) * 16];
        v16h a1 = *(const v16h*)&buf[((wm * 2 + 1) * 32 + lane) * 16];
        v16h b0 = *(const v16h*)(bpack + ((size_t)(kt * 32 + nt0    ) * 32 + lane) * 16);
        v16h b1 = *(const v16h*)(bpack + ((size_t)(kt * 32 + nt0 + 1) * 32 + lane) * 16);

        acc00 = __builtin_amdgcn_wmma_f32_16x16x32_f16(false, a0, false, b0, (short)0, acc00, false, false);
        acc01 = __builtin_amdgcn_wmma_f32_16x16x32_f16(false, a0, false, b1, (short)0, acc01, false, false);
        acc10 = __builtin_amdgcn_wmma_f32_16x16x32_f16(false, a1, false, b0, (short)0, acc10, false, false);
        acc11 = __builtin_amdgcn_wmma_f32_16x16x32_f16(false, a1, false, b1, (short)0, acc11, false, false);
        __syncthreads();                                   // all reads done before reuse
    }
#undef ISSUE

    // C/D layout: lane<16 -> M=v, N=lane ; lane>=16 -> M=8+v, N=lane-16
    const int nlane = lane & 15;
    const int mhalf = lane >> 4;
    int gn0 = N0 + (wn * 2 + 0) * 16 + nlane;
    int gn1 = N0 + (wn * 2 + 1) * 16 + nlane;
    float bias0 = bf[gn0], bias1 = bf[gn1];
    int row0 = M0 + (wm * 2 + 0) * 16 + 8 * mhalf;
    int row1 = M0 + (wm * 2 + 1) * 16 + 8 * mhalf;
#pragma unroll
    for (int v = 0; v < 8; ++v) {
        featOut[(size_t)(row0 + v) * 512 + gn0] = (h16)(acc00[v] + bias0);
        featOut[(size_t)(row0 + v) * 512 + gn1] = (h16)(acc01[v] + bias1);
        featOut[(size_t)(row1 + v) * 512 + gn0] = (h16)(acc10[v] + bias0);
        featOut[(size_t)(row1 + v) * 512 + gn1] = (h16)(acc11[v] + bias1);
    }
}

// ------------------------------------------- logits = feat(f16) @ wc + bc
__global__ __launch_bounds__(256) void logits_k(const h16* __restrict__ feat,
                                                const float* __restrict__ wc,
                                                const float* __restrict__ bc,
                                                float* __restrict__ out) {
    __shared__ float wcs[512 * CLS];
    int tid = threadIdx.x;
    for (int k = tid; k < 512 * CLS; k += 256) wcs[k] = wc[k];
    __syncthreads();
    int gm = blockIdx.x * 256 + tid;
    const float4* row = (const float4*)(feat + (size_t)gm * 512);
    float acc[CLS] = {};
    for (int kk = 0; kk < 64; ++kk) {
        float4 f4 = row[kk];
        const h16* h = (const h16*)&f4;
#pragma unroll
        for (int u = 0; u < 8; ++u) {
            float xv = (float)h[u];
            int k = kk * 8 + u;
#pragma unroll
            for (int c = 0; c < CLS; ++c) acc[c] += xv * wcs[k * CLS + c];
        }
    }
#pragma unroll
    for (int c = 0; c < CLS; ++c) out[(size_t)gm * CLS + c] = acc[c] + bc[c];
}

// ============================================================================
extern "C" void kernel_launch(void* const* d_in, const int* in_sizes, int n_in,
                              void* d_out, int out_size, void* d_ws, size_t ws_size,
                              hipStream_t stream) {
    (void)in_sizes; (void)n_in; (void)out_size; (void)ws_size;
    const int*   x    = (const int*)d_in[0];
    const float* mask = (const float*)d_in[2];
    const float* gen  = (const float*)d_in[3];
    const float* dom  = (const float*)d_in[4];
    const float* w1 = (const float*)d_in[5];  const float* b1 = (const float*)d_in[6];
    const float* w2 = (const float*)d_in[7];  const float* b2 = (const float*)d_in[8];
    const float* w3 = (const float*)d_in[9];  const float* b3 = (const float*)d_in[10];
    const float* w4 = (const float*)d_in[11]; const float* b4 = (const float*)d_in[12];
    const float* w5 = (const float*)d_in[13]; const float* b5 = (const float*)d_in[14];
    const float* wq = (const float*)d_in[15]; const float* bq = (const float*)d_in[16];
    const float* wv = (const float*)d_in[17]; const float* bv = (const float*)d_in[18];
    const float* vv = (const float*)d_in[19];
    const float* wf = (const float*)d_in[20]; const float* bf = (const float*)d_in[21];
    const float* wc = (const float*)d_in[22]; const float* bc = (const float*)d_in[23];
    float* out = (float*)d_out;

    char* ws = (char*)d_ws;
    size_t off = 0;
    auto take = [&](size_t bytes) {
        char* p = ws + off;
        off += (bytes + 255) & ~(size_t)255;
        return p;
    };
    float* xe   = (float*)take((size_t)BATCH * INCH * LSEQ * 4);
    float* xcA  = (float*)take((size_t)BATCH * CNN * LSEQ * 4);
    float* xcB  = (float*)take((size_t)BATCH * CNN * LSEQ * 4);
    float* xcT  = (float*)take((size_t)BATCH * LSEQ * CNN * 4);
    float* xcF  = (float*)take((size_t)BATCH * LSEQ * CNN * 4);
    h16*   xcH  = (h16*)  take((size_t)BATCH * LSEQ * CNN * 2);
    float* qb   = (float*)take((size_t)BATCH * LSEQ * ATT * 4);
    float* vb   = (float*)take((size_t)BATCH * LSEQ * ATT * 4);
    float* sc   = (float*)take((size_t)MROWS * 4);
    float* ci   = (float*)take((size_t)BATCH * LSEQ * CLS * 4);
    float* cj   = (float*)take((size_t)BATCH * LSEQ * CLS * 4);
    float* lmb  = (float*)take((size_t)BATCH * LSEQ * CLS * 4);
    h16*   bpk  = (h16*)  take((size_t)KTILES * 32 * 32 * 16 * 2);
    h16*   tailb= (h16*)  take((size_t)MROWS * 32 * 2);
    h16*   featA= (h16*)  take((size_t)MROWS * 512 * 2);
    h16*   featB= (h16*)  take((size_t)MROWS * 512 * 2);

    const int T = 256;
    embed_k  <<<(BATCH*INCH*LSEQ + T-1)/T, T, 0, stream>>>(x, gen, dom, xe);
    conv12_k <<<(BATCH*CNN*LSEQ  + T-1)/T, T, 0, stream>>>(xe, w1, b1, w2, b2, xcA);
    conv256_k<<<(BATCH*CNN*LSEQ  + T-1)/T, T, 0, stream>>>(xcA, w3, b3, xcB);
    conv256_k<<<(BATCH*CNN*LSEQ  + T-1)/T, T, 0, stream>>>(xcB, w4, b4, xcA);
    conv256_k<<<(BATCH*CNN*LSEQ  + T-1)/T, T, 0, stream>>>(xcA, w5, b5, xcB);
    transpose_k<<<(BATCH*LSEQ*CNN + T-1)/T, T, 0, stream>>>(xcB, xcT);
    qv_k     <<<(BATCH*LSEQ*64   + T-1)/T, T, 0, stream>>>(xcT, wq, bq, wv, bv, qb, vb);
    scores_k <<<(MROWS           + T-1)/T, T, 0, stream>>>(qb, vb, vv, mask, sc);
    attn_merge_k<<<BATCH*LSEQ, 128, 0, stream>>>(sc, xcT, mask, xcF, xcH);
    cicj_k   <<<(BATCH*LSEQ*8    + T-1)/T, T, 0, stream>>>(xcF, wc, ci, cj);
    logits0_k<<<(MROWS*8         + T-1)/T, T, 0, stream>>>(ci, cj, bc, out);
    feat0_k  <<<(MROWS*64        + T-1)/T, T, 0, stream>>>(xcH, featA);
    packwf_k <<<(KTILES*32*32    + T-1)/T, T, 0, stream>>>(wf, bpk);

    dim3 ggrid(512 / 128, MROWS / 64);           // (4, 2048)
    // ---- hop 1: probs = out (logits0)
    lm_k     <<<(BATCH*LSEQ*8  + T-1)/T, T, 0, stream>>>(out, lmb);
    tail_k   <<<(MROWS*32      + T-1)/T, T, 0, stream>>>(lmb, out, tailb);
    hop_gemm_k<<<ggrid, 256, 0, stream>>>(featA, featB, bpk, tailb, bf);
    logits_k <<<MROWS / 256, 256, 0, stream>>>(featB, wc, bc, out);
    // ---- hop 2: probs = out (logits1)
    lm_k     <<<(BATCH*LSEQ*8  + T-1)/T, T, 0, stream>>>(out, lmb);
    tail_k   <<<(MROWS*32      + T-1)/T, T, 0, stream>>>(lmb, out, tailb);
    hop_gemm_k<<<ggrid, 256, 0, stream>>>(featB, featA, bpk, tailb, bf);
    logits_k <<<MROWS / 256, 256, 0, stream>>>(featA, wc, bc, out);
}
